// NMCL2_7533372637822
// MI455X (gfx1250) — matrix operand order repaired and checked by reading, not compiled, not run on hardware
//
#include <hip/hip_runtime.h>
#include <hip/hip_bf16.h>

typedef unsigned short u16;
typedef unsigned int   u32;

typedef __attribute__((ext_vector_type(8)))  __bf16 v8bf;
typedef __attribute__((ext_vector_type(16))) __bf16 v16bf;
typedef __attribute__((ext_vector_type(8)))  float  v8f;
typedef __attribute__((ext_vector_type(4)))  u32    v4u;
typedef __attribute__((ext_vector_type(2)))  u32    v2u;

#define B_SZ  4096
#define D_SZ  1024
#define TWO_B 8192
#define KSEL  10

// GEMM tiling: 128x128 block, 8 waves in a 4x2 grid, each wave owns 32x64
#define BM   128
#define BN   128
#define BK   32
#define LDA  40      // padded LDS row stride (halfs): 80 B/row, conflict-free
#define TILE_BYTES (BM * LDA * 2)
#define CSPLIT 4
#define PSPLIT (CSPLIT * 2)   // column-split x 2 waves sharing each row
#define CCHUNK 2048  // 8192 / CSPLIT
#define NCT  (CCHUNK / BN)

// workspace layout (bytes)
#define OFF_HBF 0ull                         // 8192*1024*2 = 16 MiB bf16 h
#define OFF_PM  16777216ull                  // 4096 f32 row max of P
#define OFF_PA  (OFF_PM + 16384ull)          // 4096 i32 row argmax of P
#define OFF_DOT (OFF_PA + 16384ull)          // 4096 f32 exact dot(h_i[r], h_j[r])
#define OFF_ADJ (OFF_DOT + 16384ull)         // 4096*128 u32 adjacency bitmask (2 MiB)
#define OFF_PMS (OFF_ADJ + 2097152ull)       // PSPLIT*8192 f32 partial lse max
#define OFF_PSS (OFF_PMS + 262144ull)        // PSPLIT*8192 f32 partial lse sum
#define OFF_BS  (OFF_PSS + 262144ull)        // 32 f32 block sums

#if defined(__HIP_DEVICE_COMPILE__) && \
    __has_builtin(__builtin_amdgcn_tensor_load_to_lds) && \
    __has_builtin(__builtin_amdgcn_s_wait_tensorcnt)
#define USE_TDM 1
#else
#define USE_TDM 0
#endif

__device__ __forceinline__ u16 f2bf(float f) {
  union { float f; u32 u; } v; v.f = f;
  u32 r = v.u + 0x7FFFu + ((v.u >> 16) & 1u);   // round-to-nearest-even
  return (u16)(r >> 16);
}

#if USE_TDM
typedef __attribute__((ext_vector_type(4))) unsigned int u32x4;
typedef __attribute__((ext_vector_type(8))) int          i32x8;
typedef __attribute__((ext_vector_type(4))) int          i32x4;

__device__ __forceinline__ unsigned lds_off(const void* p) {
  return (unsigned)(unsigned long long)
      (__attribute__((address_space(3))) const void*)p;
}

// Issue one TDM load: 128 rows x 32 halfs from hbf[. , kk] into LDS, with
// 4 DWORDs of padding every 16 DWORDs -> LDS row stride = 80 B (= LDA halfs).
__device__ __forceinline__ void tdm_tile(const u16* g, unsigned ldsByteOff) {
  unsigned long long ga = (unsigned long long)(size_t)g;
  u32x4 g0 = { 1u,                                   // count=1, no gather/iterate
               ldsByteOff,                           // lds_addr [63:32]
               (u32)ga,                              // global_addr [95:64]
               (u32)(ga >> 32) | 0x80000000u };      // addr hi | type=2 [127:126]
  i32x8 g1 = { (int)((1u << 16) |                    // data_size = 2 bytes
                     (1u << 20) |                    // pad_enable
                     (3u << 22) |                    // pad_interval = 16 DWORDs
                     (3u << 25)),                    // pad_amount   = 4 DWORDs
               (int)(1024u << 16),                   // tensor_dim0 = 1024 (lo16)
               (int)(8192u << 16),                   // dim0 hi | tensor_dim1 = 8192 (lo16)
               (int)(32u << 16),                     // dim1 hi | tile_dim0 = 32
               128,                                  // tile_dim1 = 128, tile_dim2 = 0
               1024,                                 // tensor_dim0_stride = 1024
               0, 0 };                               // dim1_stride unused (2D tile)
  i32x4 gz4 = { 0, 0, 0, 0 };                        // D# groups 2/3: unused (2D)
  i32x8 gz8 = { 0, 0, 0, 0, 0, 0, 0, 0 };           // trailing group (clang-23 form)
  __builtin_amdgcn_tensor_load_to_lds(g0, g1, gz4, gz4, gz8, 0);
}
#endif

// ---------------------------------------------------------------- h -> bf16
__global__ __launch_bounds__(256) void k_convert(const float* __restrict__ hi,
                                                 const float* __restrict__ hj,
                                                 u16* __restrict__ hbf) {
  for (size_t idx = (size_t)blockIdx.x * 256 + threadIdx.x;
       idx < (size_t)TWO_B * D_SZ; idx += (size_t)gridDim.x * 256) {
    size_t r = idx >> 10, k = idx & 1023;
    float x = (r < B_SZ) ? hi[r * D_SZ + k] : hj[(r - B_SZ) * D_SZ + k];
    hbf[idx] = f2bf(x);
  }
}

// ------------------------------------------------- per-row max/argmax of P
__global__ __launch_bounds__(256) void k_rowmax(const float* __restrict__ P,
                                                float* __restrict__ Pm,
                                                int* __restrict__ Pa) {
  const int lane = threadIdx.x & 31;
  const int row  = blockIdx.x * 8 + (threadIdx.x >> 5);
  float best = -__builtin_inff(); int bidx = 0;
  #pragma unroll
  for (int t = 0; t < 4; ++t) {
    int c = lane + 32 * t;
    float v = P[(size_t)row * 128 + c];
    if (v > best) { best = v; bidx = c; }
  }
  #pragma unroll
  for (int off = 16; off; off >>= 1) {
    float ob = __shfl_xor(best, off, 32);
    int   oi = __shfl_xor(bidx, off, 32);
    if (ob > best || (ob == best && oi < bidx)) { best = ob; bidx = oi; }
  }
  if (lane == 0) { Pm[row] = best; Pa[row] = bidx; }
}

// ------------------------------------ exact f32 pos: dot(h_i[r], h_j[r])
__global__ __launch_bounds__(256) void k_rowdot(const float* __restrict__ hi,
                                                const float* __restrict__ hj,
                                                float* __restrict__ dotij) {
  const int lane = threadIdx.x & 31;
  const int row  = blockIdx.x * 8 + (threadIdx.x >> 5);
  const float* a = hi + (size_t)row * D_SZ;
  const float* b = hj + (size_t)row * D_SZ;
  float acc = 0.f;
  #pragma unroll 4
  for (int t = 0; t < D_SZ / 32; ++t) { int k = lane + 32 * t; acc = fmaf(a[k], b[k], acc); }
  #pragma unroll
  for (int off = 16; off; off >>= 1) acc += __shfl_xor(acc, off, 32);
  if (lane == 0) dotij[row] = acc;
}

// ---------------- adjacency: S row, 10th-largest threshold, bitmask output
__global__ __launch_bounds__(256) void k_adj(const float* __restrict__ Pm,
                                             const int* __restrict__ Pa,
                                             const float* __restrict__ noise,
                                             u32* __restrict__ adjbits) {
  __shared__ float Srow[B_SZ];
  __shared__ float cand[256 * KSEL];
  const int tid = threadIdx.x;
  const int i   = blockIdx.x;
  const float mi = Pm[i];
  const int   ai = Pa[i];

  float top[KSEL];
  #pragma unroll
  for (int o = 0; o < KSEL; ++o) top[o] = -__builtin_inff();

  for (int t = 0; t < B_SZ / 256; ++t) {
    int j = tid + 256 * t;
    float mj = Pm[j]; int aj = Pa[j];
    float d2 = (aj == ai) ? (mi - mj) * (mi - mj) : fmaf(mi, mi, mj * mj);
    float s  = -(sqrtf(d2) + 1e-4f * noise[(size_t)i * B_SZ + j] - ((j == i) ? 1.0f : 0.0f));
    Srow[j] = s;
    if (s > top[KSEL - 1]) {          // sorted insert (descending)
      int p = KSEL - 1;
      while (p > 0 && top[p - 1] < s) { top[p] = top[p - 1]; --p; }
      top[p] = s;
    }
  }
  #pragma unroll
  for (int o = 0; o < KSEL; ++o) cand[tid * KSEL + o] = top[o];

  // tree-merge 256 sorted top-10 lists -> global top-10 in cand[0..9]
  for (int st = 1; st < 256; st <<= 1) {
    __syncthreads();
    if ((tid & (2 * st - 1)) == 0) {
      float mrg[KSEL]; int p = 0, q = 0;
      #pragma unroll
      for (int o = 0; o < KSEL; ++o) {
        float x = cand[tid * KSEL + p], y = cand[(tid + st) * KSEL + q];
        if (x >= y) { mrg[o] = x; ++p; } else { mrg[o] = y; ++q; }
      }
      #pragma unroll
      for (int o = 0; o < KSEL; ++o) cand[tid * KSEL + o] = mrg[o];
    }
  }
  __syncthreads();
  const float kth = cand[KSEL - 1];

  if (tid < 128) {
    u32 w = 0;
    #pragma unroll
    for (int b = 0; b < 32; ++b) w |= (Srow[tid * 32 + b] >= kth) ? (1u << b) : 0u;
    adjbits[(size_t)i * 128 + tid] = w;
  }
}

// --------- fused bf16-WMMA GEMM (sim = 2*h h^T) + masked online logsumexp
__global__ __launch_bounds__(256) void k_gemm_lse(const u16* __restrict__ hbf,
                                                  const u32* __restrict__ adjbits,
                                                  float* __restrict__ pm,
                                                  float* __restrict__ ps) {
  __shared__ __attribute__((aligned(16))) u16 As[2][BM * LDA];
  __shared__ __attribute__((aligned(16))) u16 Bs[2][BN * LDA];
  const int tid  = threadIdx.x;
  const int lane = tid & 31;
  const int w    = tid >> 5;        // wave id 0..7
  const int wr   = w >> 1;          // wave row  0..3 -> rows [wr*32, wr*32+32)
  const int wc   = w & 1;           // wave col  0..1 -> cols [wc*64, wc*64+64)
  const int hh   = lane >> 4;       // half-wave
  const int ln16 = lane & 15;       // A: M index, B/C: N index
  const int rb0  = blockIdx.x * BM;
  const int cs   = blockIdx.y;

#if USE_TDM
  const unsigned asOff = lds_off(&As[0][0]);
  const unsigned bsOff = lds_off(&Bs[0][0]);
#define STAGE(kks, b) do {                                                  \
    if (w == 0) {                                                           \
      tdm_tile(hbf + (size_t)rb0 * D_SZ + (kks), asOff + (b) * TILE_BYTES); \
      tdm_tile(hbf + (size_t)cb0 * D_SZ + (kks), bsOff + (b) * TILE_BYTES); \
    } } while (0)
#else
#define STAGE(kks, b) do {                                                  \
    _Pragma("unroll")                                                       \
    for (int it = 0; it < 2; ++it) {                                        \
      int chunk = tid + 256 * it;                                           \
      int rr = chunk >> 2;                                                  \
      int ko = (chunk & 3) * 8;                                             \
      *reinterpret_cast<v4u*>(&As[b][rr * LDA + ko]) =                      \
          *reinterpret_cast<const v4u*>(&hbf[(size_t)(rb0 + rr) * D_SZ + (kks) + ko]); \
      *reinterpret_cast<v4u*>(&Bs[b][rr * LDA + ko]) =                      \
          *reinterpret_cast<const v4u*>(&hbf[(size_t)(cb0 + rr) * D_SZ + (kks) + ko]); \
    } } while (0)
#endif

  float rm[2][8], rs[2][8];
  #pragma unroll
  for (int mi = 0; mi < 2; ++mi)
    #pragma unroll
    for (int v = 0; v < 8; ++v) { rm[mi][v] = -__builtin_inff(); rs[mi][v] = 0.f; }

  for (int ct = 0; ct < NCT; ++ct) {
    const int cb0 = cs * CCHUNK + ct * BN;
    v8f acc[2][4];
    #pragma unroll
    for (int mi = 0; mi < 2; ++mi)
      #pragma unroll
      for (int f = 0; f < 4; ++f)
        #pragma unroll
        for (int v = 0; v < 8; ++v) acc[mi][f][v] = 0.0f;

    STAGE(0, 0);                       // prologue: DMA first k-slab into buf 0
    int cur = 0;
    for (int kk = 0; kk < D_SZ; kk += BK) {
#if USE_TDM
      if (w == 0) __builtin_amdgcn_s_wait_tensorcnt(0);  // buf[cur] landed
#endif
      __syncthreads();                 // publish buf[cur]; buf[cur^1] free
      if (kk + BK < D_SZ) STAGE(kk + BK, cur ^ 1);

      // A fragments (16x32 bf16): half 0 -> K {0..7,16..23}; half 1 -> {8..15,24..31}
      v16bf afrag[2];
      #pragma unroll
      for (int mi = 0; mi < 2; ++mi) {
        const u16* ar = &As[cur][(wr * 32 + mi * 16 + ln16) * LDA];
        v8bf a0 = *reinterpret_cast<const v8bf*>(ar + hh * 8);
        v8bf a1 = *reinterpret_cast<const v8bf*>(ar + 16 + hh * 8);
        afrag[mi] = __builtin_shufflevector(a0, a1, 0,1,2,3,4,5,6,7,8,9,10,11,12,13,14,15);
      }
      #pragma unroll
      for (int f = 0; f < 4; ++f) {
        // B fragment (32x16 bf16): half 0 -> K {0..7,16..23}; half 1 -> {8..15,24..31}
        const u16* br = &Bs[cur][(wc * 64 + f * 16 + ln16) * LDA + hh * 8];
        v8bf b0 = *reinterpret_cast<const v8bf*>(br);
        v8bf b1 = *reinterpret_cast<const v8bf*>(br + 16);
        v16bf bf_ = __builtin_shufflevector(b0, b1, 0,1,2,3,4,5,6,7,8,9,10,11,12,13,14,15);
        acc[0][f] = __builtin_amdgcn_wmma_f32_16x16x32_bf16(
            false, afrag[0], false, bf_, (short)0, acc[0][f], false, false);
        acc[1][f] = __builtin_amdgcn_wmma_f32_16x16x32_bf16(
            false, afrag[1], false, bf_, (short)0, acc[1][f], false, false);
      }
      cur ^= 1;
    }

    // apply adjacency mask + online (max,sum) logsumexp update
    const int cbm = cb0 & (B_SZ - 1);
    #pragma unroll
    for (int mi = 0; mi < 2; ++mi) {
      #pragma unroll
      for (int v = 0; v < 8; ++v) {
        const int r   = rb0 + wr * 32 + mi * 16 + (hh ? v + 8 : v);
        const int rbm = r & (B_SZ - 1);
        v2u aw2 = *reinterpret_cast<const v2u*>(
            &adjbits[(size_t)rbm * 128 + (cbm >> 5) + wc * 2]);
        #pragma unroll
        for (int f = 0; f < 4; ++f) {
          int nloc = f * 16 + ln16;                  // 0..63 within wave's cols
          bool masked = (aw2[nloc >> 5] >> (nloc & 31)) & 1u;
          float x = 2.0f * acc[mi][f][v];            // temperature 0.5
          if (!masked) {
            float nmax = fmaxf(rm[mi][v], x);
            rs[mi][v] = rs[mi][v] * __expf(rm[mi][v] - nmax) + __expf(x - nmax);
            rm[mi][v] = nmax;
          }
        }
      }
    }
  }
#undef STAGE

  // combine (m,s) across the 16 lanes sharing each output row
  #pragma unroll
  for (int mi = 0; mi < 2; ++mi) {
    #pragma unroll
    for (int v = 0; v < 8; ++v) {
      float m_ = rm[mi][v], s_ = rs[mi][v];
      #pragma unroll
      for (int off = 1; off < 16; off <<= 1) {
        float om = __shfl_xor(m_, off, 16);
        float os = __shfl_xor(s_, off, 16);
        float nmax = fmaxf(m_, om);
        s_ = s_ * __expf(m_ - nmax) + os * __expf(om - nmax);
        m_ = nmax;
      }
      if (ln16 == 0) {
        int r = rb0 + wr * 32 + mi * 16 + (hh ? v + 8 : v);
        int part = cs * 2 + wc;
        pm[(size_t)part * TWO_B + r] = m_;
        ps[(size_t)part * TWO_B + r] = s_;
      }
    }
  }
}

// ------------- combine split partials + pos; per-block partial sums
__global__ __launch_bounds__(256) void k_finalize(const float* __restrict__ pm,
                                                  const float* __restrict__ ps,
                                                  const float* __restrict__ dotij,
                                                  float* __restrict__ bsum) {
  __shared__ float red[256];
  const int r = blockIdx.x * 256 + threadIdx.x;
  float M = -__builtin_inff(), S = 0.f;
  #pragma unroll
  for (int cs = 0; cs < PSPLIT; ++cs) {
    float m2 = pm[(size_t)cs * TWO_B + r], s2 = ps[(size_t)cs * TWO_B + r];
    float nmax = fmaxf(M, m2);
    S = S * expf(M - nmax) + s2 * expf(m2 - nmax);
    M = nmax;
  }
  float p = 2.0f * dotij[r & (B_SZ - 1)];     // logits column 0
  float nmax = fmaxf(M, p);
  S = S * expf(M - nmax) + expf(p - nmax);
  M = nmax;
  float loss = M + logf(S) - p;

  red[threadIdx.x] = loss;
  __syncthreads();
  for (int st = 128; st; st >>= 1) {
    if (threadIdx.x < st) red[threadIdx.x] += red[threadIdx.x + st];
    __syncthreads();
  }
  if (threadIdx.x == 0) bsum[blockIdx.x] = red[0];
}

__global__ void k_finalsum(const float* __restrict__ bsum, float* __restrict__ out) {
  float v = bsum[threadIdx.x];                // 32 lanes, fixed order -> deterministic
  #pragma unroll
  for (int off = 16; off; off >>= 1) v += __shfl_xor(v, off, 32);
  if (threadIdx.x == 0) out[0] = v / (float)TWO_B;
}

extern "C" void kernel_launch(void* const* d_in, const int* in_sizes, int n_in,
                              void* d_out, int out_size, void* d_ws, size_t ws_size,
                              hipStream_t stream) {
  const float* h_i   = (const float*)d_in[0];
  const float* h_j   = (const float*)d_in[1];
  const float* P     = (const float*)d_in[2];
  const float* noise = (const float*)d_in[3];
  float* out = (float*)d_out;

  char* ws = (char*)d_ws;
  u16*   hbf     = (u16*)(ws + OFF_HBF);
  float* Pm      = (float*)(ws + OFF_PM);
  int*   Pa      = (int*)(ws + OFF_PA);
  float* dotij   = (float*)(ws + OFF_DOT);
  u32*   adjbits = (u32*)(ws + OFF_ADJ);
  float* pm      = (float*)(ws + OFF_PMS);
  float* ps      = (float*)(ws + OFF_PSS);
  float* bsum    = (float*)(ws + OFF_BS);

  k_convert<<<4096, 256, 0, stream>>>(h_i, h_j, hbf);
  k_rowmax <<<B_SZ / 8, 256, 0, stream>>>(P, Pm, Pa);
  k_rowdot <<<B_SZ / 8, 256, 0, stream>>>(h_i, h_j, dotij);
  k_adj    <<<B_SZ, 256, 0, stream>>>(Pm, Pa, noise, adjbits);

  dim3 grid(TWO_B / BM, CSPLIT);
  k_gemm_lse<<<grid, 256, 0, stream>>>(hbf, adjbits, pm, ps);

  k_finalize<<<TWO_B / 256, 256, 0, stream>>>(pm, ps, dotij, bsum);
  k_finalsum<<<1, 32, 0, stream>>>(bsum, out);
}
